// CktGNN_25769803776287
// MI455X (gfx1250) — matrix operand order repaired and checked by reading, not compiled, not run on hardware
//
#include <hip/hip_runtime.h>
#include <hip/hip_fp16.h>

typedef __attribute__((ext_vector_type(16))) _Float16 v16h;
typedef __attribute__((ext_vector_type(8)))  float    v8f;
typedef __attribute__((ext_vector_type(4)))  int      v4i_;
typedef __attribute__((address_space(1))) v4i_  as1_v4i;
typedef __attribute__((address_space(3))) v4i_  as3_v4i;

#define B_      4096
#define MAXN_   10
#define NVT_    26
#define MAXPOS_ 9
#define HS_     301
#define NZ_     56
#define EMB_    16
#define FEAT_   8
#define VS_     310     // HS + MAXPOS
#define XDIM_   35      // NVT + MAXPOS
#define GS_     309     // HS + FEAT

// padded dims
#define KPAD    320     // K = HS padded to multiple of 32
#define N1PAD   960     // 3*HS = 903 -> 960 (mult of 64)
#define N2PAD   640     // 2*HS = 602 -> 640 (mult of 64)
#define HSP4    304     // HS padded to mult of 16 for f32 strides

// ---- workspace layout (bytes) ----
#define OFF_WHH   ((size_t)0)
#define OFF_WGM   (OFF_WHH   + (size_t)N1PAD*KPAD*2)
#define OFF_HH    (OFF_WGM   + (size_t)N2PAD*KPAD*2)
#define OFF_GATED (OFF_HH    + (size_t)B_*MAXN_*KPAD*2)
#define OFF_HINH  (OFF_GATED + (size_t)B_*MAXN_*HSP4*2)
#define OFF_HINF  (OFF_HINH  + (size_t)B_*KPAD*2)
#define OFF_GH    (OFF_HINF  + (size_t)B_*HSP4*4)
#define OFF_GM2   (OFF_GH    + (size_t)B_*N1PAD*4)
#define OFF_HD    (OFF_GM2   + (size_t)B_*N2PAD*4)

union F16x16 { v16h h; uint4 q[2]; };

__device__ __forceinline__ float sigmoidf_(float x){ return 1.f/(1.f + __expf(-x)); }
__device__ __forceinline__ int clampn_(int n){ return min(max(n,1),MAXN_); }

// CDNA5 async global->LDS copy path (ASYNCcnt), if the toolchain exposes it.
#if __has_builtin(__builtin_amdgcn_global_load_async_to_lds_b128)
#define ASYNC_LDS 1
#endif

__device__ __forceinline__ void wait_async0(){
#if defined(ASYNC_LDS)
#if __has_builtin(__builtin_amdgcn_s_wait_asynccnt)
    __builtin_amdgcn_s_wait_asynccnt(0);
#else
    asm volatile("s_wait_asynccnt 0x0" ::: "memory");
#endif
#endif
}

__device__ __forceinline__ void async_cp16(const _Float16* src, _Float16* lds_dst){
#if defined(ASYNC_LDS)
    __builtin_amdgcn_global_load_async_to_lds_b128(
        (as1_v4i*)(v4i_*)src, (as3_v4i*)(v4i_*)lds_dst, 0, 0);
#endif
}

// stage one 128x32 A tile + 64x32 B tile (f16) into LDS; 3 x 16B per thread
__device__ __forceinline__ void stage_tile(
    const _Float16* __restrict__ A, int lda,
    const _Float16* __restrict__ Bw,
    int mBase, int nBase, int kt,
    _Float16* a_lds, _Float16* b_lds, int tid)
{
    const int r = tid >> 2, c = (tid & 3) * 8;
    const _Float16* gA0 = &A[(size_t)(mBase + r)      * lda  + kt + c];
    const _Float16* gA1 = &A[(size_t)(mBase + r + 64) * lda  + kt + c];
    const _Float16* gB  = &Bw[(size_t)(nBase + r)     * KPAD + kt + c];
#if defined(ASYNC_LDS)
    async_cp16(gA0, &a_lds[r*40 + c]);
    async_cp16(gA1, &a_lds[(r+64)*40 + c]);
    async_cp16(gB,  &b_lds[r*40 + c]);
#else
    uint4 x0 = *(const uint4*)gA0;      // three independent loads -> one wait
    uint4 x1 = *(const uint4*)gA1;
    uint4 y  = *(const uint4*)gB;
    *(uint4*)&a_lds[r*40 + c]      = x0;
    *(uint4*)&a_lds[(r+64)*40 + c] = x1;
    *(uint4*)&b_lds[r*40 + c]      = y;
#endif
}

// ---------------------------------------------------------------------------
// 0. one-time weight prep: fp32 -> zero-padded f16 [Npad x KPAD], row = out col
// ---------------------------------------------------------------------------
__global__ __launch_bounds__(256) void prep_weights(
    const float* __restrict__ Whh, const float* __restrict__ Wg,
    const float* __restrict__ Wm,
    _Float16* __restrict__ WhhH, _Float16* __restrict__ WgmH)
{
    int idx = blockIdx.x * blockDim.x + threadIdx.x;
    const int tot1 = N1PAD * KPAD;
    if (idx < tot1) {
        int n = idx / KPAD, k = idx % KPAD;
        float v = (n < 3*HS_ && k < HS_) ? Whh[n*HS_ + k] : 0.f;
        WhhH[idx] = (_Float16)v;
    } else {
        int j = idx - tot1;
        if (j < N2PAD * KPAD) {
            int n = j / KPAD, k = j % KPAD;
            float v = 0.f;
            if (k < HS_) {
                if (n < HS_)        v = Wg[n*VS_ + k];
                else if (n < 2*HS_) v = Wm[(n-HS_)*VS_ + k];
            }
            WgmH[j] = (_Float16)v;
        }
    }
}

// ---------------------------------------------------------------------------
// 1. predecessor aggregation: Hin[b] = sum_{u<v} A[b,u,v] * gated[b,u]
// ---------------------------------------------------------------------------
__global__ __launch_bounds__(256) void aggregate(
    const int* __restrict__ adj, const int* __restrict__ vcount,
    const _Float16* __restrict__ gated,
    float* __restrict__ HinF, _Float16* __restrict__ HinH, int v)
{
    int idx = blockIdx.x * blockDim.x + threadIdx.x;   // B_*KPAD threads
    int b = idx / KPAD, j = idx % KPAD;
    int nb = clampn_(vcount[b]);
    float s = 0.f;
    if (j < HS_ && v < nb) {
        const int* arow = adj + b*MAXN_*MAXN_;
        const _Float16* g = gated + (size_t)b*MAXN_*HSP4 + j;
        for (int u = 0; u < v; ++u)
            if (u < nb && arow[u*MAXN_ + v]) s += (float)g[u*HSP4];
    }
    if (j < HSP4) HinF[(size_t)b*HSP4 + j] = s;
    HinH[(size_t)b*KPAD + j] = (_Float16)s;
}

// ---------------------------------------------------------------------------
// 2. WMMA GEMM: C[4096 x ldc] = A[4096 x K] @ Bw.T (+bias), f16 in / f32 acc
//    Bw is pre-padded f16 [Npad x KPAD] (row n = output column n).
//    Block tile 128(M) x 64(N), 8 waves, wave = 16x64 (4 WMMA tiles).
//    Double-buffered LDS; next k-tile streams (async if available) while
//    the current one feeds the WMMA pipe. One barrier per k-tile.
// ---------------------------------------------------------------------------
__global__ __launch_bounds__(256) void gemm_f16(
    const _Float16* __restrict__ A, int lda,
    const _Float16* __restrict__ Bw,
    float* __restrict__ C, int ldc, int Nreal,
    const float* __restrict__ bias)
{
    __shared__ _Float16 sA[2][128*40];   // 128 rows, stride 40 halves
    __shared__ _Float16 sB[2][64*40];    // 64 cols (n-major), stride 40 halves

    const int tid  = threadIdx.x;
    const int wave = tid >> 5, lane = tid & 31;
    const int hlf  = lane >> 4, l15 = lane & 15;
    const int mBase = blockIdx.y * 128;
    const int nBase = blockIdx.x * 64;

    v8f acc[4] = {};

    stage_tile(A, lda, Bw, mBase, nBase, 0, sA[0], sB[0], tid);
    int cur = 0;

    for (int kt = 0; kt < KPAD; kt += 32) {
        wait_async0();           // (async path) our tile-kt writes landed
        __syncthreads();         // everyone's tile-kt visible; prev reads done

        if (kt + 32 < KPAD)      // prefetch next tile into the other buffer
            stage_tile(A, lda, Bw, mBase, nBase, kt + 32,
                       sA[cur ^ 1], sB[cur ^ 1], tid);

        // A fragment per ISA layout: lane(row=l15, half) holds
        // K = {8*hlf..+8} and {16+8*hlf..+8}
        F16x16 af;
        const _Float16* ap = &sA[cur][(wave*16 + l15)*40];
        af.q[0] = *(const uint4*)(ap + hlf*8);
        af.q[1] = *(const uint4*)(ap + 16 + hlf*8);

        #pragma unroll
        for (int s = 0; s < 4; ++s) {
            // B fragment: lane(col=l15) holds K = {8*hlf..+8} and {16+8*hlf..+8}
            F16x16 bf;
            const _Float16* bp = &sB[cur][(s*16 + l15)*40 + hlf*8];
            bf.q[0] = *(const uint4*)(bp);
            bf.q[1] = *(const uint4*)(bp + 16);
            acc[s] = __builtin_amdgcn_wmma_f32_16x16x32_f16(
                false, af.h, false, bf.h, (short)0, acc[s], false, false);
        }
        cur ^= 1;
    }

    // epilogue: C layout — lanes 0-15 rows 0..7 in vgpr j, lanes 16-31 rows 8..15
    const int mRow0 = mBase + wave*16 + hlf*8;
    #pragma unroll
    for (int s = 0; s < 4; ++s) {
        int col = nBase + s*16 + l15;
        float bval = (bias != nullptr && col < Nreal) ? bias[col] : 0.f;
        #pragma unroll
        for (int j = 0; j < 8; ++j)
            C[(size_t)(mRow0 + j)*ldc + col] = acc[s][j] + bval;
    }
}

// ---------------------------------------------------------------------------
// 3. GRU elementwise (one-hot x => column gathers from W_ih)
// ---------------------------------------------------------------------------
__global__ __launch_bounds__(256) void gru_elem(
    const int* __restrict__ node_type, const int* __restrict__ pos,
    const int* __restrict__ vcount,
    const float* __restrict__ Wih, const float* __restrict__ bih,
    const float* __restrict__ bhh,
    const float* __restrict__ gh, const float* __restrict__ HinF,
    _Float16* __restrict__ Hh, int v)
{
    int idx = blockIdx.x * blockDim.x + threadIdx.x;   // B_*KPAD threads
    int b = idx / KPAD, j = idx % KPAD;
    float out = 0.f;
    int nb = clampn_(vcount[b]);
    if (j < HS_ && v < nb) {
        int t = node_type[b*MAXN_ + v];
        int p = pos[b*MAXN_ + v];
        float ir = Wih[j*XDIM_ + t]          + Wih[j*XDIM_ + NVT_ + p]          + bih[j];
        float iz = Wih[(j+HS_)*XDIM_ + t]    + Wih[(j+HS_)*XDIM_ + NVT_ + p]    + bih[j+HS_];
        float in_= Wih[(j+2*HS_)*XDIM_ + t]  + Wih[(j+2*HS_)*XDIM_ + NVT_ + p]  + bih[j+2*HS_];
        float hr, hz, hn, h;
        if (v == 0) { hr = bhh[j]; hz = bhh[j+HS_]; hn = bhh[j+2*HS_]; h = 0.f; }
        else {
            const float* g = gh + (size_t)b*N1PAD;
            hr = g[j]; hz = g[j+HS_]; hn = g[j+2*HS_];
            h  = HinF[(size_t)b*HSP4 + j];
        }
        float rg = sigmoidf_(ir + hr);
        float zg = sigmoidf_(iz + hz);
        float ng = tanhf(in_ + rg*hn);
        out = (1.f - zg)*ng + zg*h;
    }
    Hh[((size_t)b*MAXN_ + v)*KPAD + j] = (_Float16)out;
}

// ---------------------------------------------------------------------------
// 4. gated = sigmoid(g-part + bg + Wg[:,HS+p]) * (m-part + Wm[:,HS+p])
// ---------------------------------------------------------------------------
__global__ __launch_bounds__(256) void gated_epi(
    const int* __restrict__ pos, const int* __restrict__ vcount,
    const float* __restrict__ gm2,
    const float* __restrict__ Wg, const float* __restrict__ bg,
    const float* __restrict__ Wm,
    _Float16* __restrict__ gated, int v)
{
    int idx = blockIdx.x * blockDim.x + threadIdx.x;   // B_*HSP4 threads
    int b = idx / HSP4, j = idx % HSP4;
    float out = 0.f;
    int nb = clampn_(vcount[b]);
    if (j < HS_ && v < nb) {
        int p = pos[b*MAXN_ + v];
        const float* c = gm2 + (size_t)b*N2PAD;
        float g = c[j]       + bg[j] + Wg[j*VS_ + HS_ + p];
        float m = c[HS_ + j]         + Wm[j*VS_ + HS_ + p];
        out = sigmoidf_(g) * m;
    }
    gated[((size_t)b*MAXN_ + v)*HSP4 + j] = (_Float16)out;
}

// ---------------------------------------------------------------------------
// 5. df feature + df_enc MLP (27 -> 16 -> 8), one thread per graph
// ---------------------------------------------------------------------------
__global__ __launch_bounds__(256) void df_enc(
    const int* __restrict__ pos, const int* __restrict__ vcount,
    const float* __restrict__ rr, const float* __restrict__ cc,
    const float* __restrict__ gmv,
    const float* __restrict__ W1, const float* __restrict__ b1,
    const float* __restrict__ W2, const float* __restrict__ b2,
    float* __restrict__ Hd)
{
    int b = blockIdx.x * blockDim.x + threadIdx.x;
    if (b >= B_) return;
    int nb = clampn_(vcount[b]);
    float df[3*MAXPOS_];
    #pragma unroll
    for (int i = 0; i < 3*MAXPOS_; ++i) df[i] = 0.f;
    for (int v = 0; v < MAXN_; ++v) {
        if (v < nb) {
            int p = pos[b*MAXN_ + v] * 3;
            df[p]   = rr[b*MAXN_ + v];
            df[p+1] = cc[b*MAXN_ + v];
            df[p+2] = gmv[b*MAXN_ + v];
        }
    }
    float e[EMB_];
    for (int i = 0; i < EMB_; ++i) {
        float s = b1[i];
        for (int k = 0; k < 3*MAXPOS_; ++k) s += W1[i*(3*MAXPOS_) + k] * df[k];
        e[i] = fmaxf(s, 0.f);
    }
    for (int f = 0; f < FEAT_; ++f) {
        float s = b2[f];
        for (int k = 0; k < EMB_; ++k) s += W2[f*EMB_ + k] * e[k];
        Hd[b*FEAT_ + f] = s;
    }
}

// ---------------------------------------------------------------------------
// 6. heads: Hfull = [H[b, n-1], Hd[b]] ; mu/logvar = Hfull @ W.T + b
// ---------------------------------------------------------------------------
__global__ __launch_bounds__(128) void heads(
    const _Float16* __restrict__ Hh, const float* __restrict__ Hd,
    const int* __restrict__ vcount,
    const float* __restrict__ Wmu, const float* __restrict__ bmu,
    const float* __restrict__ Wlv, const float* __restrict__ blv,
    float* __restrict__ out)
{
    __shared__ float hf[GS_];
    int b = blockIdx.x;
    int nb = clampn_(vcount[b]);
    const _Float16* hg = Hh + ((size_t)b*MAXN_ + (nb-1))*KPAD;
    for (int k = threadIdx.x; k < GS_; k += 128)
        hf[k] = (k < HS_) ? (float)hg[k] : Hd[b*FEAT_ + (k - HS_)];
    __syncthreads();
    int o = threadIdx.x;
    if (o < 2*NZ_) {
        const float* Wrow = (o < NZ_) ? (Wmu + (size_t)o*GS_) : (Wlv + (size_t)(o-NZ_)*GS_);
        float s = (o < NZ_) ? bmu[o] : blv[o-NZ_];
        for (int k = 0; k < GS_; ++k) s += Wrow[k]*hf[k];
        size_t oi = (o < NZ_) ? ((size_t)b*NZ_ + o)
                              : ((size_t)B_*NZ_ + (size_t)b*NZ_ + (o-NZ_));
        out[oi] = s;
    }
}

// ---------------------------------------------------------------------------
extern "C" void kernel_launch(void* const* d_in, const int* in_sizes, int n_in,
                              void* d_out, int out_size, void* d_ws, size_t ws_size,
                              hipStream_t stream)
{
    const int*   node_type = (const int*)  d_in[0];
    const int*   pos       = (const int*)  d_in[1];
    const int*   adj       = (const int*)  d_in[2];
    const int*   vcount    = (const int*)  d_in[3];
    const float* rr        = (const float*)d_in[4];
    const float* cc        = (const float*)d_in[5];
    const float* gmv       = (const float*)d_in[6];
    const float* W_ih      = (const float*)d_in[7];
    const float* W_hh      = (const float*)d_in[8];
    const float* b_ih      = (const float*)d_in[9];
    const float* b_hh      = (const float*)d_in[10];
    const float* Wg        = (const float*)d_in[11];
    const float* bg        = (const float*)d_in[12];
    const float* Wm        = (const float*)d_in[13];
    const float* W1        = (const float*)d_in[14];
    const float* b1        = (const float*)d_in[15];
    const float* W2        = (const float*)d_in[16];
    const float* b2        = (const float*)d_in[17];
    const float* Wmu       = (const float*)d_in[18];
    const float* bmu       = (const float*)d_in[19];
    const float* Wlv       = (const float*)d_in[20];
    const float* blv       = (const float*)d_in[21];

    char* ws = (char*)d_ws;
    _Float16* WhhH  = (_Float16*)(ws + OFF_WHH);
    _Float16* WgmH  = (_Float16*)(ws + OFF_WGM);
    _Float16* Hh    = (_Float16*)(ws + OFF_HH);
    _Float16* gated = (_Float16*)(ws + OFF_GATED);
    _Float16* HinH  = (_Float16*)(ws + OFF_HINH);
    float*    HinF  = (float*)   (ws + OFF_HINF);
    float*    gh    = (float*)   (ws + OFF_GH);
    float*    gm2   = (float*)   (ws + OFF_GM2);
    float*    Hd    = (float*)   (ws + OFF_HD);

    const dim3 blk256(256), blk128(128);
    const int gBK   = (B_*KPAD)/256;   // 5120
    const int gBH   = (B_*HSP4)/256;   // 4864
    const dim3 gG1(N1PAD/64, B_/128);  // (15, 32)
    const dim3 gG2(N2PAD/64, B_/128);  // (10, 32)

    prep_weights<<<(N1PAD*KPAD + N2PAD*KPAD)/256, blk256, 0, stream>>>(
        W_hh, Wg, Wm, WhhH, WgmH);

    // step v = 0 : Hin = 0 -> GRU uses b_hh directly; then gated[:,0]
    gru_elem<<<gBK, blk256, 0, stream>>>(node_type, pos, vcount,
        W_ih, b_ih, b_hh, gh, HinF, Hh, 0);
    gemm_f16<<<gG2, blk256, 0, stream>>>(Hh /* plane 0 */, MAXN_*KPAD,
        WgmH, gm2, N2PAD, 2*HS_, nullptr);
    gated_epi<<<gBH, blk256, 0, stream>>>(pos, vcount, gm2, Wg, bg, Wm, gated, 0);

    for (int v = 1; v < MAXN_; ++v) {
        aggregate<<<gBK, blk256, 0, stream>>>(adj, vcount, gated, HinF, HinH, v);
        gemm_f16<<<gG1, blk256, 0, stream>>>(HinH, KPAD, WhhH, gh, N1PAD, 3*HS_, b_hh);
        gru_elem<<<gBK, blk256, 0, stream>>>(node_type, pos, vcount,
            W_ih, b_ih, b_hh, gh, HinF, Hh, v);
        if (v < MAXN_ - 1) {
            gemm_f16<<<gG2, blk256, 0, stream>>>(Hh + (size_t)v*KPAD, MAXN_*KPAD,
                WgmH, gm2, N2PAD, 2*HS_, nullptr);
            gated_epi<<<gBH, blk256, 0, stream>>>(pos, vcount, gm2, Wg, bg, Wm, gated, v);
        }
    }

    df_enc<<<B_/256, blk256, 0, stream>>>(pos, vcount, rr, cc, gmv,
        W1, b1, W2, b2, Hd);
    heads<<<B_, blk128, 0, stream>>>(Hh, Hd, vcount, Wmu, bmu, Wlv, blv,
        (float*)d_out);
}
